// VRNN_Gauss_54331336295047
// MI455X (gfx1250) — matrix operand order repaired and checked by reading, not compiled, not run on hardware
//
#include <hip/hip_runtime.h>
#include <hip/hip_bf16.h>

// ---------------------------------------------------------------------------
// VRNN forward for MI455X (gfx1250): bf16 WMMA GEMMs, weights pre-packed into
// a contiguous-k B-fragment lane layout (same k order as A), split-A concat GEMMs,
// dual-output (mean/logvar) epilogues, deterministic loss reduction.
// GEMM: software-pipelined with ping-pong B registers (unroll-by-2 so no
// register shuffles feed the WMMAs), branch-free steady state, templated on
// A-source mode and full-tile-N so hot instantiations carry no bounds checks.
// ---------------------------------------------------------------------------

typedef __attribute__((ext_vector_type(16))) __bf16 v16bf;
typedef __attribute__((ext_vector_type(8)))  float  v8f;

struct Bytes32 { uint4 lo, hi; };   // 32 bytes == one v16bf fragment

__device__ __forceinline__ unsigned short f2bf_rne(float f) {
  unsigned int u = __float_as_uint(f);
  unsigned int r = u + 0x7FFFu + ((u >> 16) & 1u);   // round-to-nearest-even
  return (unsigned short)(r >> 16);
}

// Two fp32 -> packed bf16x2 with ONE v_perm_b32 (truncation rounding):
// result[15:0] = a[31:16], result[31:16] = b[31:16]
__device__ __forceinline__ unsigned int pack2bf(float a, float b) {
  return __builtin_amdgcn_perm(__float_as_uint(b), __float_as_uint(a), 0x07060302u);
}

#define TILE_M 128
#define TILE_N 128
#define LDS_STRIDE 40   // 32 bf16 + 8 pad (80B rows -> conflict-free b128 reads)

#define NT 300
#define NB 256
#define ND 76
#define NH 1024
#define NZ 128
#define LOG2PI_F 1.8378770664093453f

// ---------------------------------------------------------------------------
// Weight packing: fp32 [K,N] (optionally two sources side by side, zero pad)
// -> bf16 blocks of 512 elems per (16-col, 32-k) tile in contiguous-k B lane order (matches A):
// elem(k,n) -> blk=(n/16)*(Kp/32)+(k/32), lane=(n&15)+16*((k>>4)&1), slot k&15.
// Each wave then fetches a B fragment as 2 contiguous global uint4 loads.
// ---------------------------------------------------------------------------
__global__ void pack_weights_kernel(const float* __restrict__ W1, int K1, int N1,
                                    const float* __restrict__ W2, int K2, int N2,
                                    unsigned short* __restrict__ dst,
                                    int Kp, int Np) {
  long long idx = (long long)blockIdx.x * 256 + threadIdx.x;
  long long total = (long long)Kp * Np;
  if (idx >= total) return;
  int k = (int)(idx / Np);
  int n = (int)(idx % Np);
  float v = 0.f;
  if (n < N1) {
    if (k < K1) v = W1[(long long)k * N1 + n];
  } else if (n - N1 < N2) {
    if (k < K2) v = W2[(long long)k * N2 + (n - N1)];
  }
  long long blk = (long long)(n >> 4) * (Kp >> 5) + (k >> 5);
  int lane = (n & 15) + ((k & 16) ? 16 : 0);
  int slot = k & 15;
  dst[blk * 512 + lane * 16 + slot] = f2bf_rne(v);
}

// ---------------------------------------------------------------------------
// GEMM: C = act(concat(A0,A1) @ W + bias), A fp32 (converted to bf16 in LDS),
// W pre-packed bf16.  Block = 128x128x(K step 32), 8 waves (2x4), each wave
// holds 4x2 16x16 f32 accumulators -> 8 WMMAs per K-chunk, B reused over M.
// MODE: 0 = single source, Kp==K0 (no checks; aligned float4 loads)
//       1 = split concat, Kp==K0+K1, K0%16==0 (one pointer select per group)
//       2 = general (per-element bounds; only used for the 76-wide phi_y in)
// FULLN: Np is a multiple of TILE_N -> B loads need no column guard.
// ---------------------------------------------------------------------------
template <int ACT, int MODE, bool FULLN>
__global__ __launch_bounds__(256) void gemm_wmma_kernel(
    const float* __restrict__ A0, int lda0, int K0,
    const float* __restrict__ A1, int lda1, int K1,
    int Kp, int Np,
    const unsigned short* __restrict__ Wp,
    const float* __restrict__ bias1, float* __restrict__ C1, int ldc1, int N1,
    const float* __restrict__ bias2, float* __restrict__ C2, int ldc2, int N2) {
  __shared__ __align__(16) unsigned short sA[TILE_M * LDS_STRIDE];

  const int tid  = threadIdx.x;
  const int lane = tid & 31;
  const int wave = tid >> 5;
  const int wm   = wave >> 2;            // 0..1 : 64-row strip
  const int wn   = wave & 3;             // 0..3 : 32-col strip
  const int mBase = blockIdx.y * TILE_M;
  const int nBase = blockIdx.x * TILE_N;
  const int Ktot = K0 + K1;
  const int kBlocks = Kp >> 5;

  v8f zeroAcc = {0.f, 0.f, 0.f, 0.f, 0.f, 0.f, 0.f, 0.f};
  v8f acc[4][2];
#pragma unroll
  for (int i = 0; i < 4; ++i)
#pragma unroll
    for (int j = 0; j < 2; ++j) acc[i][j] = zeroAcc;

  // A-tile staging: each thread owns one (row, 16-wide k group)
  const int rowA = tid >> 1;
  const int cgA  = (tid & 1) << 4;
  const int mA   = mBase + rowA;
  uint4* sDst = (uint4*)&sA[rowA * LDS_STRIDE + cgA];

  float4 aReg[4];
  // fetch one 16-float k-group into 4 float4 regs (mode-specialized)
  auto loadA = [&](int kt) {
    const int kbase = kt + cgA;
    if (MODE == 0) {
      const float4* p = (const float4*)(A0 + mA * lda0 + kbase);
#pragma unroll
      for (int i = 0; i < 4; ++i) aReg[i] = p[i];
    } else if (MODE == 1) {
      const float* base = (kbase < K0) ? (A0 + mA * lda0 + kbase)
                                       : (A1 + mA * lda1 + (kbase - K0));
      const float4* p = (const float4*)base;
#pragma unroll
      for (int i = 0; i < 4; ++i) aReg[i] = p[i];
    } else {
      float t[16];
#pragma unroll
      for (int i = 0; i < 16; ++i) {
        const int k = kbase + i;
        float v = 0.f;
        if (k < K0)        v = A0[mA * lda0 + k];
        else if (k < Ktot) v = A1[mA * lda1 + (k - K0)];
        t[i] = v;
      }
#pragma unroll
      for (int i = 0; i < 4; ++i)
        aReg[i] = make_float4(t[4*i], t[4*i+1], t[4*i+2], t[4*i+3]);
    }
  };

  // stage prefetched A regs into LDS (8 v_perm + 2 ds_store_b128)
  auto stageA = [&]() {
    uint4 q0, q1;
    q0.x = pack2bf(aReg[0].x, aReg[0].y); q0.y = pack2bf(aReg[0].z, aReg[0].w);
    q0.z = pack2bf(aReg[1].x, aReg[1].y); q0.w = pack2bf(aReg[1].z, aReg[1].w);
    q1.x = pack2bf(aReg[2].x, aReg[2].y); q1.y = pack2bf(aReg[2].z, aReg[2].w);
    q1.z = pack2bf(aReg[3].x, aReg[3].y); q1.w = pack2bf(aReg[3].z, aReg[3].w);
    sDst[0] = q0;
    sDst[1] = q1;
  };

  // B fragment pointers (wave-uniform validity; weights stay L2-resident)
  const int nc0 = nBase + wn * 32;
  const int nc1 = nc0 + 16;
  const bool nv0 = nc0 < Np;
  const bool nv1 = nc1 < Np;
  const uint4* bp0 = (const uint4*)Wp + (long long)(nc0 >> 4) * kBlocks * 64 + lane * 2;
  const uint4* bp1 = (const uint4*)Wp + (long long)(nc1 >> 4) * kBlocks * 64 + lane * 2;

  auto loadB = [&](int kb, Bytes32& b0, Bytes32& b1) {
    if (FULLN) {
      const uint4* p0 = bp0 + (long long)kb * 64;
      const uint4* p1 = bp1 + (long long)kb * 64;
      b0.lo = p0[0]; b0.hi = p0[1];
      b1.lo = p1[0]; b1.hi = p1[1];
    } else {
      const uint4 zq = make_uint4(0, 0, 0, 0);
      b0.lo = zq; b0.hi = zq; b1.lo = zq; b1.hi = zq;
      if (nv0) { const uint4* p = bp0 + (long long)kb * 64; b0.lo = p[0]; b0.hi = p[1]; }
      if (nv1) { const uint4* p = bp1 + (long long)kb * 64; b1.lo = p[0]; b1.hi = p[1]; }
    }
  };

  const int halfOff = (lane >> 4) << 4;  // ushort offset for A frag halves
  const int rowSel  = lane & 15;
  const unsigned short* aBase = &sA[(wm * 64 + rowSel) * LDS_STRIDE + halfOff];

  // A fragments from LDS + 8 WMMAs (B reused across 4 M-fragments)
  auto mmaChunk = [&](const Bytes32& rb0, const Bytes32& rb1) {
    v16bf b0 = __builtin_bit_cast(v16bf, rb0);
    v16bf b1 = __builtin_bit_cast(v16bf, rb1);
#pragma unroll
    for (int mf = 0; mf < 4; ++mf) {
      Bytes32 raw;
      raw.lo = *(const uint4*)(aBase + mf * 16 * LDS_STRIDE);
      raw.hi = *(const uint4*)(aBase + mf * 16 * LDS_STRIDE + 8);
      v16bf afrag = __builtin_bit_cast(v16bf, raw);
      acc[mf][0] = __builtin_amdgcn_wmma_f32_16x16x32_bf16(
          false, afrag, false, b0, (short)0, acc[mf][0], false, false);
      acc[mf][1] = __builtin_amdgcn_wmma_f32_16x16x32_bf16(
          false, afrag, false, b1, (short)0, acc[mf][1], false, false);
    }
  };

  // ---- software pipeline, unrolled by 2 with ping-pong B register sets so
  //      the WMMAs consume the registers the loads wrote (no shuffles).
  Bytes32 bA0, bA1, bB0, bB1;
  loadA(0);
  loadB(0, bA0, bA1);
  int kb = 0;
  for (; kb + 2 < kBlocks; kb += 2) {
    // even chunk: consume bA, prefetch kb+1 into bB
    __syncthreads();
    stageA();
    __syncthreads();
    loadA((kb + 1) << 5);
    loadB(kb + 1, bB0, bB1);
    mmaChunk(bA0, bA1);
    // odd chunk: consume bB, prefetch kb+2 into bA
    __syncthreads();
    stageA();
    __syncthreads();
    loadA((kb + 2) << 5);
    loadB(kb + 2, bA0, bA1);
    mmaChunk(bB0, bB1);
  }
  if (kb + 2 == kBlocks) {
    // two chunks left
    __syncthreads();
    stageA();
    __syncthreads();
    loadA((kb + 1) << 5);
    loadB(kb + 1, bB0, bB1);
    mmaChunk(bA0, bA1);
    __syncthreads();
    stageA();
    __syncthreads();
    mmaChunk(bB0, bB1);
  } else {
    // one chunk left
    __syncthreads();
    stageA();
    __syncthreads();
    mmaChunk(bA0, bA1);
  }

  // ---- epilogue: C frag layout = lane 0-15 col, vgpr r row (+8 for hi lanes)
  const int colLane = lane & 15;
  const int rowOff  = (lane >> 4) << 3;
#pragma unroll
  for (int mf = 0; mf < 4; ++mf) {
    const int mrow = mBase + wm * 64 + mf * 16 + rowOff;
#pragma unroll
    for (int nf = 0; nf < 2; ++nf) {
      const int col = nBase + wn * 32 + nf * 16 + colLane;
      if (col < N1) {
        const float b = bias1 ? bias1[col] : 0.f;
#pragma unroll
        for (int r2 = 0; r2 < 8; ++r2) {
          float v = acc[mf][nf][r2] + b;
          if (ACT) v = fmaxf(v, 0.f);
          C1[(mrow + r2) * ldc1 + col] = v;
        }
      } else if (col - N1 < N2) {
        const int c2 = col - N1;
        const float b = bias2 ? bias2[c2] : 0.f;
#pragma unroll
        for (int r2 = 0; r2 < 8; ++r2) {
          float v = fmaxf(acc[mf][nf][r2] + b, 0.f);   // logvar head: ReLU
          C2[(mrow + r2) * ldc2 + c2] = v;
        }
      }
    }
  }
}

// ---------------------------------------------------------------------------
// Elementwise step kernels
// ---------------------------------------------------------------------------
__global__ void zsample_kernel(const float* __restrict__ em,
                               const float* __restrict__ elv,
                               const float* __restrict__ eps,
                               float* __restrict__ z, float* __restrict__ zout,
                               int n) {
  int i = blockIdx.x * 256 + threadIdx.x;
  if (i >= n) return;
  float zv = em[i] + __expf(0.5f * elv[i]) * eps[i];
  z[i] = zv;
  zout[i] = zv;
}

__global__ void gru_kernel(const float* __restrict__ gi,
                           const float* __restrict__ gh,
                           float* __restrict__ h, float* __restrict__ hout,
                           int n) {
  int i = blockIdx.x * 256 + threadIdx.x;
  if (i >= n) return;
  int m = i / NH, j = i % NH;
  const float* gim = gi + (long long)m * 3 * NH;
  const float* ghm = gh + (long long)m * 3 * NH;
  float r  = 1.f / (1.f + __expf(-(gim[j] + ghm[j])));
  float zg = 1.f / (1.f + __expf(-(gim[NH + j] + ghm[NH + j])));
  float nn = tanhf(gim[2 * NH + j] + r * ghm[2 * NH + j]);
  float hv = (1.f - zg) * nn + zg * h[i];
  h[i] = hv;
  hout[i] = hv;
}

__global__ void kld_kernel(const float* __restrict__ em, const float* __restrict__ elv,
                           const float* __restrict__ pm, const float* __restrict__ plv,
                           float* __restrict__ kv, float* __restrict__ partial, int n) {
  __shared__ float red[256];
  int i = blockIdx.x * 256 + threadIdx.x;
  float e = 0.f;
  if (i < n) {
    float d = em[i] - pm[i];
    e = (plv[i] - elv[i] - 1.f) + (__expf(elv[i]) + d * d) * __expf(-plv[i]);
    kv[i] = e;
  }
  red[threadIdx.x] = e;
  __syncthreads();
  for (int s = 128; s > 0; s >>= 1) {
    if (threadIdx.x < s) red[threadIdx.x] += red[threadIdx.x + s];
    __syncthreads();
  }
  if (threadIdx.x == 0) partial[blockIdx.x] = 0.5f * red[0];
}

__global__ void rec_kernel(const float* __restrict__ y, const float* __restrict__ dm,
                           const float* __restrict__ dlv, float* __restrict__ partial,
                           int n) {
  __shared__ float red[256];
  int i = blockIdx.x * 256 + threadIdx.x;
  float e = 0.f;
  if (i < n) {
    float lv = dlv[i];
    float d = y[i] - dm[i];
    e = 0.5f * LOG2PI_F + 0.5f * lv + d * d * 0.5f * __expf(-lv);  // -logp
  }
  red[threadIdx.x] = e;
  __syncthreads();
  for (int s = 128; s > 0; s >>= 1) {
    if (threadIdx.x < s) red[threadIdx.x] += red[threadIdx.x + s];
    __syncthreads();
  }
  if (threadIdx.x == 0) partial[blockIdx.x] = red[0];
}

__global__ void reduce_losses_kernel(const float* __restrict__ prec, int nrec,
                                     const float* __restrict__ pkld, int nkld,
                                     float* __restrict__ out) {
  __shared__ float red[256];
  float s = 0.f;
  for (int i = threadIdx.x; i < nrec; i += 256) s += prec[i];
  red[threadIdx.x] = s;
  __syncthreads();
  for (int st = 128; st > 0; st >>= 1) {
    if (threadIdx.x < st) red[threadIdx.x] += red[threadIdx.x + st];
    __syncthreads();
  }
  if (threadIdx.x == 0) out[0] = red[0];
  __syncthreads();
  s = 0.f;
  for (int i = threadIdx.x; i < nkld; i += 256) s += pkld[i];
  red[threadIdx.x] = s;
  __syncthreads();
  for (int st = 128; st > 0; st >>= 1) {
    if (threadIdx.x < st) red[threadIdx.x] += red[threadIdx.x + st];
    __syncthreads();
  }
  if (threadIdx.x == 0) out[1] = red[0];
}

__global__ void copy_kernel(const float* __restrict__ src, float* __restrict__ dst,
                            long long n) {
  long long i = (long long)blockIdx.x * 256 + threadIdx.x;
  if (i < n) dst[i] = src[i];
}

__global__ void zero_kernel(float* __restrict__ dst, long long n) {
  long long i = (long long)blockIdx.x * 256 + threadIdx.x;
  if (i < n) dst[i] = 0.f;
}

// ---------------------------------------------------------------------------
// Host-side helpers
// ---------------------------------------------------------------------------
static inline void gemmL(hipStream_t s, int act, int mode,
                         const float* A0, int lda0, int K0,
                         const float* A1, int lda1, int K1,
                         int Kp, int Np, const unsigned short* Wp,
                         const float* b1, float* C1, int ldc1, int N1,
                         const float* b2, float* C2, int ldc2, int N2) {
  dim3 grid((Np + TILE_N - 1) / TILE_N, NB / TILE_M);
  const bool fulln = (Np % TILE_N) == 0;
#define GEMM_CALL(A, M, F)                                                     \
  gemm_wmma_kernel<A, M, F><<<grid, 256, 0, s>>>(A0, lda0, K0, A1, lda1, K1,   \
                                                 Kp, Np, Wp, b1, C1, ldc1, N1, \
                                                 b2, C2, ldc2, N2)
  if (act) {
    if (mode == 0) { if (fulln) GEMM_CALL(1, 0, true); else GEMM_CALL(1, 0, false); }
    else if (mode == 1) { if (fulln) GEMM_CALL(1, 1, true); else GEMM_CALL(1, 1, false); }
    else { if (fulln) GEMM_CALL(1, 2, true); else GEMM_CALL(1, 2, false); }
  } else {
    if (mode == 0) { if (fulln) GEMM_CALL(0, 0, true); else GEMM_CALL(0, 0, false); }
    else if (mode == 1) { if (fulln) GEMM_CALL(0, 1, true); else GEMM_CALL(0, 1, false); }
    else { if (fulln) GEMM_CALL(0, 2, true); else GEMM_CALL(0, 2, false); }
  }
#undef GEMM_CALL
}

static inline void packL(hipStream_t s, const float* W1, int K1, int N1,
                         const float* W2, int K2, int N2,
                         unsigned short* dst, int Kp, int Np) {
  long long total = (long long)Kp * Np;
  int blocks = (int)((total + 255) / 256);
  pack_weights_kernel<<<blocks, 256, 0, s>>>(W1, K1, N1, W2, K2, N2, dst, Kp, Np);
}

extern "C" void kernel_launch(void* const* d_in, const int* in_sizes, int n_in,
                              void* d_out, int out_size, void* d_ws, size_t ws_size,
                              hipStream_t stream) {
  (void)in_sizes; (void)n_in; (void)out_size; (void)ws_size;
  // Input order (jax pytree: y, eps, then params dict with sorted keys; tuples
  // in (W1,b1,W2,b2)/(W,b) order):
  const float* y_all   = (const float*)d_in[0];   // [T,B,76]
  const float* eps_all = (const float*)d_in[1];   // [T,B,128]
  const float* decW1   = (const float*)d_in[2];   const float* decB1 = (const float*)d_in[3];
  const float* decW2   = (const float*)d_in[4];   const float* decB2 = (const float*)d_in[5];
  const float* dlvW    = (const float*)d_in[6];   const float* dlvB  = (const float*)d_in[7];
  const float* dmW1    = (const float*)d_in[8];   const float* dmB1  = (const float*)d_in[9];
  const float* dmW2    = (const float*)d_in[10];  const float* dmB2  = (const float*)d_in[11];
  const float* encW1   = (const float*)d_in[12];  const float* encB1 = (const float*)d_in[13];
  const float* encW2   = (const float*)d_in[14];  const float* encB2 = (const float*)d_in[15];
  const float* elvW    = (const float*)d_in[16];  const float* elvB  = (const float*)d_in[17];
  const float* emW     = (const float*)d_in[18];  const float* emB   = (const float*)d_in[19];
  const float* Whh     = (const float*)d_in[20];  // [1024,3072]
  const float* Wih     = (const float*)d_in[21];  // [2048,3072]
  const float* pyW1    = (const float*)d_in[22];  const float* pyB1  = (const float*)d_in[23];
  const float* pyW2    = (const float*)d_in[24];  const float* pyB2  = (const float*)d_in[25];
  const float* pzW1    = (const float*)d_in[26];  const float* pzB1  = (const float*)d_in[27];
  const float* pzW2    = (const float*)d_in[28];  const float* pzB2  = (const float*)d_in[29];
  const float* prW1    = (const float*)d_in[30];  const float* prB1  = (const float*)d_in[31];
  const float* prW2    = (const float*)d_in[32];  const float* prB2  = (const float*)d_in[33];
  const float* plvW    = (const float*)d_in[34];  const float* plvB  = (const float*)d_in[35];
  const float* pmW     = (const float*)d_in[36];  const float* pmB   = (const float*)d_in[37];

  // Output slices (tuple concatenated flat, all fp32)
  const long long sBD = (long long)NB * ND, sBZ = (long long)NB * NZ, sBH = (long long)NB * NH;
  float* out   = (float*)d_out;
  float* o_y   = out + 2;
  float* o_em  = o_y   + (long long)NT * sBD;
  float* o_elv = o_em  + (long long)NT * sBZ;
  float* o_dm  = o_elv + (long long)NT * sBZ;
  float* o_dlv = o_dm  + (long long)NT * sBD;
  float* o_kv  = o_dlv + (long long)NT * sBD;
  float* o_z   = o_kv  + (long long)NT * sBZ;
  float* o_h   = o_z   + (long long)NT * sBZ;

  // Workspace bump allocator (packed bf16 weights ~44MB + fp32 activations ~15MB)
  char* wsb = (char*)d_ws;
  size_t off = 0;
  auto alloc = [&](size_t bytes) -> void* {
    void* p = wsb + off;
    off = (off + bytes + 255) & ~(size_t)255;
    return p;
  };
  auto allocU = [&](long long elems) { return (unsigned short*)alloc((size_t)elems * 2); };
  auto allocF = [&](long long elems) { return (float*)alloc((size_t)elems * 4); };

  unsigned short* P_py1  = allocU(96   * 1024);
  unsigned short* P_py2  = allocU(1024 * 1024);
  unsigned short* P_enc1 = allocU(2048 * 1024);
  unsigned short* P_enc2 = allocU(1024 * 1024);
  unsigned short* P_eml  = allocU(1024 * 256);
  unsigned short* P_pr1  = allocU(1024 * 1024);
  unsigned short* P_pr2  = allocU(1024 * 1024);
  unsigned short* P_pml  = allocU(1024 * 256);
  unsigned short* P_pz1  = allocU(128  * 1024);
  unsigned short* P_pz2  = allocU(1024 * 1024);
  unsigned short* P_dec1 = allocU(2048 * 1024);
  unsigned short* P_dec2 = allocU(1024 * 1024);
  unsigned short* P_dm1  = allocU(1024 * 1024);
  unsigned short* P_dm2  = allocU(1024 * 80);
  unsigned short* P_dlv  = allocU(1024 * 80);
  unsigned short* P_wih  = allocU(2048 * 3072);
  unsigned short* P_whh  = allocU(1024 * 3072);

  float* h      = allocF(sBH);
  float* tmp1   = allocF(sBH);
  float* phiy   = allocF(sBH);
  float* phiz   = allocF(sBH);
  float* enc_t  = allocF(sBH);
  float* pri_t  = allocF(sBH);
  float* dec_t  = allocF(sBH);
  float* priorm = allocF(sBZ);
  float* priorl = allocF(sBZ);
  float* zbuf   = allocF(sBZ);
  float* gibuf  = allocF((long long)NB * 3 * NH);
  float* ghbuf  = allocF((long long)NB * 3 * NH);
  float* p_kld  = allocF((long long)NT * 128);
  float* p_rec  = allocF((long long)NT * 76);

  // ---- one-time (per call) weight packing into WMMA-B layout
  packL(stream, pyW1, 76, 1024,   nullptr, 0, 0, P_py1, 96, 1024);
  packL(stream, pyW2, 1024, 1024, nullptr, 0, 0, P_py2, 1024, 1024);
  packL(stream, encW1, 2048, 1024, nullptr, 0, 0, P_enc1, 2048, 1024);
  packL(stream, encW2, 1024, 1024, nullptr, 0, 0, P_enc2, 1024, 1024);
  packL(stream, emW, 1024, 128, elvW, 1024, 128, P_eml, 1024, 256);
  packL(stream, prW1, 1024, 1024, nullptr, 0, 0, P_pr1, 1024, 1024);
  packL(stream, prW2, 1024, 1024, nullptr, 0, 0, P_pr2, 1024, 1024);
  packL(stream, pmW, 1024, 128, plvW, 1024, 128, P_pml, 1024, 256);
  packL(stream, pzW1, 128, 1024,  nullptr, 0, 0, P_pz1, 128, 1024);
  packL(stream, pzW2, 1024, 1024, nullptr, 0, 0, P_pz2, 1024, 1024);
  packL(stream, decW1, 2048, 1024, nullptr, 0, 0, P_dec1, 2048, 1024);
  packL(stream, decW2, 1024, 1024, nullptr, 0, 0, P_dec2, 1024, 1024);
  packL(stream, dmW1, 1024, 1024, nullptr, 0, 0, P_dm1, 1024, 1024);
  packL(stream, dmW2, 1024, 76,   nullptr, 0, 0, P_dm2, 1024, 80);
  packL(stream, dlvW, 1024, 76,   nullptr, 0, 0, P_dlv, 1024, 80);
  packL(stream, Wih, 2048, 3072,  nullptr, 0, 0, P_wih, 2048, 3072);
  packL(stream, Whh, 1024, 3072,  nullptr, 0, 0, P_whh, 1024, 3072);

  // h0 = 0 ; y passthrough
  zero_kernel<<<(int)((sBH + 255) / 256), 256, 0, stream>>>(h, sBH);
  {
    long long n = (long long)NT * sBD;
    copy_kernel<<<(int)((n + 255) / 256), 256, 0, stream>>>(y_all, o_y, n);
  }

  // ---- sequential scan over T steps
  for (int t = 0; t < NT; ++t) {
    const float* y_t   = y_all   + (long long)t * sBD;
    const float* eps_t = eps_all + (long long)t * sBZ;
    float* em_t  = o_em  + (long long)t * sBZ;
    float* elv_t = o_elv + (long long)t * sBZ;
    float* dm_t  = o_dm  + (long long)t * sBD;
    float* dlv_t = o_dlv + (long long)t * sBD;

    // phi_y = lin2(relu(lin1(y_t)))                       (K=76 -> general mode)
    gemmL(stream, 1, 2, y_t, ND, ND, nullptr, 0, 0, 96, 1024, P_py1, pyB1, tmp1, NH, NH, nullptr, nullptr, 0, 0);
    gemmL(stream, 0, 0, tmp1, NH, NH, nullptr, 0, 0, 1024, 1024, P_py2, pyB2, phiy, NH, NH, nullptr, nullptr, 0, 0);
    // enc_t = relu(lin2(relu(lin1(concat(phi_y, h)))))    (split mode)
    gemmL(stream, 1, 1, phiy, NH, NH, h, NH, NH, 2048, 1024, P_enc1, encB1, tmp1, NH, NH, nullptr, nullptr, 0, 0);
    gemmL(stream, 1, 0, tmp1, NH, NH, nullptr, 0, 0, 1024, 1024, P_enc2, encB2, enc_t, NH, NH, nullptr, nullptr, 0, 0);
    // enc_mean (id) + enc_logvar (relu) fused, written straight to d_out
    gemmL(stream, 0, 0, enc_t, NH, NH, nullptr, 0, 0, 1024, 256, P_eml, emB, em_t, NZ, NZ, elvB, elv_t, NZ, NZ);
    // prior
    gemmL(stream, 1, 0, h, NH, NH, nullptr, 0, 0, 1024, 1024, P_pr1, prB1, tmp1, NH, NH, nullptr, nullptr, 0, 0);
    gemmL(stream, 0, 0, tmp1, NH, NH, nullptr, 0, 0, 1024, 1024, P_pr2, prB2, pri_t, NH, NH, nullptr, nullptr, 0, 0);
    gemmL(stream, 0, 0, pri_t, NH, NH, nullptr, 0, 0, 1024, 256, P_pml, pmB, priorm, NZ, NZ, plvB, priorl, NZ, NZ);
    // z = enc_mean + exp(0.5*enc_logvar)*eps
    zsample_kernel<<<128, 256, 0, stream>>>(em_t, elv_t, eps_t, zbuf,
                                            o_z + (long long)t * sBZ, (int)sBZ);
    // phi_z
    gemmL(stream, 1, 0, zbuf, NZ, NZ, nullptr, 0, 0, 128, 1024, P_pz1, pzB1, tmp1, NH, NH, nullptr, nullptr, 0, 0);
    gemmL(stream, 0, 0, tmp1, NH, NH, nullptr, 0, 0, 1024, 1024, P_pz2, pzB2, phiz, NH, NH, nullptr, nullptr, 0, 0);
    // dec_t = relu(lin2(relu(lin1(concat(phi_z, h)))))    (split mode)
    gemmL(stream, 1, 1, phiz, NH, NH, h, NH, NH, 2048, 1024, P_dec1, decB1, tmp1, NH, NH, nullptr, nullptr, 0, 0);
    gemmL(stream, 1, 0, tmp1, NH, NH, nullptr, 0, 0, 1024, 1024, P_dec2, decB2, dec_t, NH, NH, nullptr, nullptr, 0, 0);
    // dec_mean MLP + dec_logvar head (both to d_out)
    gemmL(stream, 1, 0, dec_t, NH, NH, nullptr, 0, 0, 1024, 1024, P_dm1, dmB1, tmp1, NH, NH, nullptr, nullptr, 0, 0);
    gemmL(stream, 0, 0, tmp1, NH, NH, nullptr, 0, 0, 1024, 80, P_dm2, dmB2, dm_t, ND, ND, nullptr, nullptr, 0, 0);
    gemmL(stream, 1, 0, dec_t, NH, NH, nullptr, 0, 0, 1024, 80, P_dlv, dlvB, dlv_t, ND, ND, nullptr, nullptr, 0, 0);
    // GRU: gi = concat(phi_y, phi_z) @ Wih ; gh = h @ Whh ; gate fuse -> h
    gemmL(stream, 0, 1, phiy, NH, NH, phiz, NH, NH, 2048, 3072, P_wih, nullptr, gibuf, 3 * NH, 3 * NH, nullptr, nullptr, 0, 0);
    gemmL(stream, 0, 0, h, NH, NH, nullptr, 0, 0, 1024, 3072, P_whh, nullptr, ghbuf, 3 * NH, 3 * NH, nullptr, nullptr, 0, 0);
    gru_kernel<<<1024, 256, 0, stream>>>(gibuf, ghbuf, h, o_h + (long long)t * sBH, (int)sBH);
    // losses (deterministic partials)
    kld_kernel<<<128, 256, 0, stream>>>(em_t, elv_t, priorm, priorl,
                                        o_kv + (long long)t * sBZ, p_kld + t * 128, (int)sBZ);
    rec_kernel<<<76, 256, 0, stream>>>(y_t, dm_t, dlv_t, p_rec + t * 76, (int)sBD);
  }

  reduce_losses_kernel<<<1, 256, 0, stream>>>(p_rec, NT * 76, p_kld, NT * 128, out);
}
